// DoubleHeadRNN_70609262346893
// MI455X (gfx1250) — matrix operand order repaired and checked by reading, not compiled, not run on hardware
//
#include <hip/hip_runtime.h>
#include <hip/hip_bf16.h>

// DoubleHeadRNN for MI455X (gfx1250, wave32, WMMA bf16 16x16x32).
// B=64, T=1024, D=512, H=1024, 3H=3072.
// Persistent kernel: 128 blocks, grid-wide atomic barrier per timestep,
// K-split x2 per output tile with LDS reduction.

#define NB 64
#define NT 1024
#define ND 512
#define NH 1024
#define N3H 3072
#define NT3H 192   // 3072/16 column tiles over 3H
#define NTH 64     // 1024/16 column tiles over H
#define KWT 16     // 512/32 K-tiles for W
#define KUT 32     // 1024/32 K-tiles for U
#define GRID_BLOCKS 128

typedef __attribute__((ext_vector_type(16))) __bf16 v16bf;
typedef __attribute__((ext_vector_type(8)))  float  v8f;

union V16U8 { v16bf v; unsigned int u[8]; };

// round-to-nearest-even fp32 -> bf16 bits
__device__ __forceinline__ unsigned short f2bf_rn(float f) {
  unsigned int u = __builtin_bit_cast(unsigned int, f);
  u += 0x7FFFu + ((u >> 16) & 1u);
  return (unsigned short)(u >> 16);
}
__device__ __forceinline__ unsigned int pack2_rn(float f0, float f1) {
  return (unsigned int)f2bf_rn(f0) | ((unsigned int)f2bf_rn(f1) << 16);
}

__device__ __forceinline__ float fast_sigmoid(float v) {
  return __builtin_amdgcn_rcpf(1.f + __expf(-v));   // v_exp + v_rcp
}
__device__ __forceinline__ float fast_tanh(float v) {
  return 2.f * fast_sigmoid(2.f * v) - 1.f;
}

// A-fragment (16x32 bf16) from a row-major bf16 row: two 16B loads, no VALU.
// lanes 0-15 -> M=lane, K = e; lanes 16-31: K = 16+e (matches B fragments).
__device__ __forceinline__ v16bf load_a_bf(const unsigned short* __restrict__ row,
                                           int k0, int lane) {
  const unsigned short* p = row + k0 + ((lane >> 4) << 4);
  uint4 c0 = *(const uint4*)p;          // 8 bf16
  uint4 c1 = *(const uint4*)(p + 8);    // 8 bf16, +8 elements
  V16U8 r;
  r.u[0] = c0.x; r.u[1] = c0.y; r.u[2] = c0.z; r.u[3] = c0.w;
  r.u[4] = c1.x; r.u[5] = c1.y; r.u[6] = c1.z; r.u[7] = c1.w;
  return r.v;
}

__device__ __forceinline__ v8f wmma_bf16(v16bf a, v16bf b, v8f c) {
  return __builtin_amdgcn_wmma_f32_16x16x32_bf16(
      false, a, false, b, (short)0, c, false, false);
}

// ---------------------------------------------------------------------------
// Prep: W/U (fp32 [K][3072]) -> bf16 WMMA B-fragments.
// Within a K=32 tile: lanes 0-15 hold K=0..15 (element e -> k=e),
// lanes 16-31 hold K=16..31; n = nt*16 + (lane&15). 32B contiguous per lane.
// ---------------------------------------------------------------------------
__global__ void prep_frags_kernel(const float* __restrict__ W0, const float* __restrict__ U0,
                                  const float* __restrict__ W1, const float* __restrict__ U1,
                                  unsigned short* __restrict__ Wf,
                                  unsigned short* __restrict__ Uf) {
  int gid = blockIdx.x * blockDim.x + threadIdx.x;
  const int WL = 2 * NT3H * KWT * 32;
  const int UL = 2 * NT3H * KUT * 32;
  if (gid < WL) {
    int lane = gid & 31; int r = gid >> 5;
    int kt = r & (KWT - 1); r >>= 4;
    int nt = r % NT3H; int head = r / NT3H;
    const float* src = head ? W1 : W0;          // [512][3072]
    int n  = nt * 16 + (lane & 15);
    int kb = kt * 32 + ((lane >> 4) << 4);
    unsigned short* dst = Wf + (size_t)gid * 16;
    #pragma unroll
    for (int e = 0; e < 16; ++e)
      dst[e] = f2bf_rn(src[(size_t)(kb + e) * N3H + n]);
  } else if (gid < WL + UL) {
    int g = gid - WL;
    int lane = g & 31; int r = g >> 5;
    int kt = r & (KUT - 1); r >>= 5;
    int nt = r % NT3H; int head = r / NT3H;
    const float* src = head ? U1 : U0;          // [1024][3072]
    int n  = nt * 16 + (lane & 15);
    int kb = kt * 32 + ((lane >> 4) << 4);
    unsigned short* dst = Uf + (size_t)g * 16;
    #pragma unroll
    for (int e = 0; e < 16; ++e)
      dst[e] = f2bf_rn(src[(size_t)(kb + e) * N3H + n]);
  }
}

// x (fp32) -> bf16 row-major copy, 8 elements per thread.
__global__ void xconv_kernel(const float* __restrict__ x,
                             unsigned short* __restrict__ xbf, int n8) {
  int i = blockIdx.x * blockDim.x + threadIdx.x;
  if (i >= n8) return;
  const float4* p = (const float4*)(x + (size_t)i * 8);
  float4 a = p[0], b = p[1];
  uint4 o;
  o.x = pack2_rn(a.x, a.y);
  o.y = pack2_rn(a.z, a.w);
  o.z = pack2_rn(b.x, b.y);
  o.w = pack2_rn(b.z, b.w);
  *(uint4*)(xbf + (size_t)i * 8) = o;
}

__global__ void zero_kernel(float* __restrict__ p, int n) {
  int i = blockIdx.x * blockDim.x + threadIdx.x;
  if (i < n) p[i] = 0.f;
}

// ---------------------------------------------------------------------------
// Persistent GRU: all 1024 timesteps for both heads in one kernel.
// 128 blocks x 8 waves. Each 16x16 h_new tile = 2 waves (K split 24/24
// k-iters), reduced through LDS. Grid-wide sense-free monotonic barrier.
// ---------------------------------------------------------------------------
__global__ void __launch_bounds__(256)
gru_persistent_kernel(const unsigned short* __restrict__ xbf,
                      const float* __restrict__ bi0, const float* __restrict__ br0,
                      const float* __restrict__ bi1, const float* __restrict__ br1,
                      const unsigned short* __restrict__ Wf,
                      const unsigned short* __restrict__ Uf,
                      float* __restrict__ h32,          // [2 buf][2][64][1024] f32
                      unsigned short* __restrict__ hbf, // [2 buf][2][64][1024] bf16
                      unsigned int* __restrict__ counter) {
  const int lane = threadIdx.x & 31;
  const int lw   = threadIdx.x >> 5;                    // 0..7
  const int tile = blockIdx.x * 4 + (lw >> 1);          // 0..511
  const int ks   = lw & 1;                              // K-split half
  const int tl   = lw >> 1;                             // tile slot in block
  const int head = tile >> 8;
  const int mt   = (tile >> 6) & 3;
  const int nc   = tile & 63;
  const int m0   = mt * 16;
  const int bidx = m0 + (lane & 15);                    // batch row for A

  __shared__ float red[4][3][8][32];                    // [tile][Z,R,HH][vgpr][lane] = 12KB

  // per-column gate biases (uniform over t)
  const int n = nc * 16 + (lane & 15);
  const float* bi = head ? bi1 : bi0;
  const float* br = head ? br1 : br0;
  const float bz  = bi[n]          + br[n];
  const float brg = bi[NH + n]     + br[NH + n];
  const float bxh = bi[2 * NH + n];
  const float bhh = br[2 * NH + n];

  const size_t hbW = (size_t)head * NT3H;
  const size_t HSZ = (size_t)2 * NB * NH;               // one ping-pong buffer

  for (int t = 0; t < NT; ++t) {
    const int cur = t & 1, nxt = cur ^ 1;
    const unsigned short* hrow = hbf + (size_t)cur * HSZ
                                     + ((size_t)head * NB + bidx) * NH;
    v8f accZ = {}, accR = {}, accXH = {}, accHH = {};

    if (ks == 0) {
      // W part (K=512) + first quarter of U
      const unsigned short* xrow = xbf + ((size_t)bidx * NT + t) * ND;
      for (int kk = 0; kk < KWT; ++kk) {
        v16bf a = load_a_bf(xrow, kk * 32, lane);
        const unsigned short* f0 = Wf + (((hbW + (0 * NTH + nc)) * KWT + kk) * 32 + lane) * 16;
        const unsigned short* f1 = Wf + (((hbW + (1 * NTH + nc)) * KWT + kk) * 32 + lane) * 16;
        const unsigned short* f2 = Wf + (((hbW + (2 * NTH + nc)) * KWT + kk) * 32 + lane) * 16;
        accZ  = wmma_bf16(a, *(const v16bf*)f0, accZ);
        accR  = wmma_bf16(a, *(const v16bf*)f1, accR);
        accXH = wmma_bf16(a, *(const v16bf*)f2, accXH);
      }
      for (int kk = 0; kk < 8; ++kk) {
        v16bf a = load_a_bf(hrow, kk * 32, lane);
        const unsigned short* f0 = Uf + (((hbW + (0 * NTH + nc)) * KUT + kk) * 32 + lane) * 16;
        const unsigned short* f1 = Uf + (((hbW + (1 * NTH + nc)) * KUT + kk) * 32 + lane) * 16;
        const unsigned short* f2 = Uf + (((hbW + (2 * NTH + nc)) * KUT + kk) * 32 + lane) * 16;
        accZ  = wmma_bf16(a, *(const v16bf*)f0, accZ);
        accR  = wmma_bf16(a, *(const v16bf*)f1, accR);
        accHH = wmma_bf16(a, *(const v16bf*)f2, accHH);
      }
    } else {
      // remaining 3/4 of U (K=768)
      for (int kk = 8; kk < KUT; ++kk) {
        v16bf a = load_a_bf(hrow, kk * 32, lane);
        const unsigned short* f0 = Uf + (((hbW + (0 * NTH + nc)) * KUT + kk) * 32 + lane) * 16;
        const unsigned short* f1 = Uf + (((hbW + (1 * NTH + nc)) * KUT + kk) * 32 + lane) * 16;
        const unsigned short* f2 = Uf + (((hbW + (2 * NTH + nc)) * KUT + kk) * 32 + lane) * 16;
        accZ  = wmma_bf16(a, *(const v16bf*)f0, accZ);
        accR  = wmma_bf16(a, *(const v16bf*)f1, accR);
        accHH = wmma_bf16(a, *(const v16bf*)f2, accHH);
      }
      #pragma unroll
      for (int i = 0; i < 8; ++i) {
        red[tl][0][i][lane] = accZ[i];
        red[tl][1][i][lane] = accR[i];
        red[tl][2][i][lane] = accHH[i];
      }
    }
    __syncthreads();

    if (ks == 0) {
      const float* h32in = h32 + (size_t)cur * HSZ;
      float* h32out      = h32 + (size_t)nxt * HSZ;
      unsigned short* hbfout = hbf + (size_t)nxt * HSZ;
      const int mrow0 = m0 + ((lane >> 4) << 3);  // C/D: vgpr i -> M=mrow0+i
      #pragma unroll
      for (int i = 0; i < 8; ++i) {
        const float z = fast_sigmoid(accZ[i] + red[tl][0][i][lane] + bz);
        const float r = fast_sigmoid(accR[i] + red[tl][1][i][lane] + brg);
        const float hh = accHH[i] + red[tl][2][i][lane] + bhh;
        const float cand = fast_tanh(accXH[i] + bxh + r * hh);
        const size_t idx = ((size_t)head * NB + (mrow0 + i)) * NH + n;
        const float hnew = z * h32in[idx] + (1.f - z) * cand;
        h32out[idx] = hnew;
        hbfout[idx] = f2bf_rn(hnew);
      }
    }

    // publish h, then grid-wide barrier (monotonic counter, agent scope)
    __threadfence();
    __syncthreads();
    if (threadIdx.x == 0) {
      __hip_atomic_fetch_add(counter, 1u, __ATOMIC_ACQ_REL, __HIP_MEMORY_SCOPE_AGENT);
      const unsigned int target = (unsigned int)(t + 1) * GRID_BLOCKS;
      while (__hip_atomic_load(counter, __ATOMIC_ACQUIRE, __HIP_MEMORY_SCOPE_AGENT) < target)
        __builtin_amdgcn_s_sleep(2);
    }
    __syncthreads();
  }
}

// ---------------------------------------------------------------------------
// Workspace layout (bytes):
//   [0,        4K ) barrier counter (zeroed each call)
//   [4K,   4K+1M ) h fp32 ping-pong  [2][2][64][1024]
//   [.. , +512K  ) h bf16 ping-pong  [2][2][64][1024]
//   [2M,      8M ) W bf16 B-fragments
//   [8M,     20M ) U bf16 B-fragments
//   [20M,    84M ) x bf16 [64][1024][512]
// ---------------------------------------------------------------------------
extern "C" void kernel_launch(void* const* d_in, const int* in_sizes, int n_in,
                              void* d_out, int out_size, void* d_ws, size_t ws_size,
                              hipStream_t stream) {
  const float* x   = (const float*)d_in[0];
  const float* W0  = (const float*)d_in[1];
  const float* U0  = (const float*)d_in[2];
  const float* bi0 = (const float*)d_in[3];
  const float* br0 = (const float*)d_in[4];
  const float* W1  = (const float*)d_in[5];
  const float* U1  = (const float*)d_in[6];
  const float* bi1 = (const float*)d_in[7];
  const float* br1 = (const float*)d_in[8];

  char* ws = (char*)d_ws;
  unsigned int*   counter = (unsigned int*)ws;
  float*          h32 = (float*)(ws + 4096);
  unsigned short* hbf = (unsigned short*)(ws + 4096 + (size_t)1024 * 1024);
  unsigned short* Wf  = (unsigned short*)(ws + (size_t)2  * 1024 * 1024);
  unsigned short* Uf  = (unsigned short*)(ws + (size_t)8  * 1024 * 1024);
  unsigned short* xbf = (unsigned short*)(ws + (size_t)20 * 1024 * 1024);

  // zero counter + both h fp32 buffers + both h bf16 buffers (contiguous)
  const int zf = (4096 + 1024 * 1024 + 512 * 1024) / 4;
  zero_kernel<<<(zf + 255) / 256, 256, 0, stream>>>((float*)ws, zf);

  const int prep_tasks = 2 * NT3H * KWT * 32 + 2 * NT3H * KUT * 32;
  prep_frags_kernel<<<(prep_tasks + 255) / 256, 256, 0, stream>>>(W0, U0, W1, U1, Wf, Uf);

  const int n8 = NB * NT * ND / 8;
  xconv_kernel<<<(n8 + 255) / 256, 256, 0, stream>>>(x, xbf, n8);

  gru_persistent_kernel<<<GRID_BLOCKS, 256, 0, stream>>>(
      xbf, bi0, br0, bi1, br1, Wf, Uf, h32, hbf, counter);

  // T=1024 even -> final state in buffer 0; layout [head][64][1024] == d_out.
  hipMemcpyAsync(d_out, h32, (size_t)2 * NB * NH * sizeof(float),
                 hipMemcpyDeviceToDevice, stream);
}
